// EquivariantStructureDenoisingModule_34471407517995
// MI455X (gfx1250) — compile-verified, matrix operand order checked by reading, not run on hardware
//
#include <hip/hip_runtime.h>
#include <math.h>
#include <stdint.h>

typedef __attribute__((ext_vector_type(16))) __bf16 v16bf;
typedef __attribute__((ext_vector_type(8)))  float  v8f;

#define FIBER    128
#define IN_DIM   1152
#define HIDDEN   1024
#define OUT_DIMS 512
#define EPS_F    1e-4f

// GEMM tiling: block tile 128x128, BK=32, 8 waves in 4(M) x 2(N), wave tile 32x64
#define BM   128
#define BN   128
#define BK   32
#define LDSK 40   // BK + 8 pad -> conflict-free 16-row fragment reads (80B row stride)

#define USE_ASYNC_STAGE 1   // gfx1250 GLOBAL_LOAD_ASYNC_TO_LDS_B128 path

// ---------------------------------------------------------------------------
// 16-byte global -> LDS copy. Async DMA (ASYNCcnt) on gfx1250, sync fallback.
// LDS byte offset = low 32 bits of generic LDS pointer (ISA §10.2 aperture).
// ---------------------------------------------------------------------------
__device__ __forceinline__ void cp16_g2l(void* lds, const void* g)
{
#if USE_ASYNC_STAGE
    const uint32_t l = (uint32_t)(uintptr_t)lds;
    asm volatile("global_load_async_to_lds_b128 %0, %1, off"
                 :: "v"(l), "v"(g) : "memory");
#else
    *(uint4*)lds = *(const uint4*)g;
#endif
}

__device__ __forceinline__ void wait_stage(bool next_in_flight)
{
#if USE_ASYNC_STAGE
    if (next_in_flight) asm volatile("s_wait_asynccnt 0x4" ::: "memory");
    else                asm volatile("s_wait_asynccnt 0x0" ::: "memory");
#else
    (void)next_in_flight;
#endif
}

// ---------------------------------------------------------------------------
// Kernel 1: build X1 = [x0 | vec_local (f*3+k) | vec_norm | extra] in bf16
// ---------------------------------------------------------------------------
__global__ __launch_bounds__(128)
void build_x1_kernel(const float* __restrict__ x, const float* __restrict__ rot,
                     const float* __restrict__ extra, __bf16* __restrict__ X1)
{
    const int n = blockIdx.x;
    const int f = threadIdx.x;               // fiber index 0..127
    __shared__ float R[9];
    if (f < 9) R[f] = rot[(size_t)n * 9 + f];
    __syncthreads();

    const float* xr = x + (size_t)n * 4 * FIBER;
    const float s0 = xr[f];
    const float v1 = xr[1 * FIBER + f];
    const float v2 = xr[2 * FIBER + f];
    const float v3 = xr[3 * FIBER + f];

    __bf16* o = X1 + (size_t)n * IN_DIM;
    o[f] = (__bf16)s0;
    #pragma unroll
    for (int k = 0; k < 3; ++k)   // vec_local[n,f,k] = sum_c vec[n,c,f]*rot[n,c,k]
        o[FIBER + f * 3 + k] = (__bf16)(v1 * R[k] + v2 * R[3 + k] + v3 * R[6 + k]);
    o[4 * FIBER + f] = (__bf16)sqrtf(v1 * v1 + v2 * v2 + v3 * v3 + EPS_F);

    const float* er = extra + (size_t)n * 512;
    #pragma unroll
    for (int j = 0; j < 4; ++j)
        o[5 * FIBER + j * FIBER + f] = (__bf16)er[j * FIBER + f];
}

// ---------------------------------------------------------------------------
// Kernel 2: W (K x Nn, row major) -> Wt (Nn x K, row major) in bf16
// ---------------------------------------------------------------------------
__global__ __launch_bounds__(256)
void transpose_bf16_kernel(const float* __restrict__ W, __bf16* __restrict__ Wt,
                           int K, int Nn)
{
    const int i = blockIdx.x * 256 + threadIdx.x;   // i indexes Wt flat [n][k]
    if (i >= K * Nn) return;
    const int n = i / K;
    const int k = i - n * K;
    Wt[i] = (__bf16)W[(size_t)k * Nn + n];
}

// ---------------------------------------------------------------------------
// Kernel 3: double-buffered bf16 WMMA GEMM.  C(MxN) = A(MxK) * Bt(NxK)^T + bias
// EPI==0 : exact GELU, store bf16.   EPI==1 : store f32.
// ---------------------------------------------------------------------------
union FragU { v16bf v; uint4 u4[2]; };

template <int EPI>
__global__ __launch_bounds__(256)
void gemm_wmma_kernel(const __bf16* __restrict__ A, const __bf16* __restrict__ Bt,
                      const float* __restrict__ bias, void* __restrict__ Cout,
                      int M, int Nn, int K)
{
    __shared__ __bf16 sA[2][BM * LDSK];
    __shared__ __bf16 sB[2][BN * LDSK];

    const int tid  = threadIdx.x;
    const int lane = tid & 31;
    const int wave = tid >> 5;       // 0..7
    const int wm   = wave >> 1;      // 0..3  (M direction, 32 rows each)
    const int wn   = wave & 1;       // 0..1  (N direction, 64 cols each)
    const int r    = lane & 15;      // row within fragment
    const int half = lane >> 4;      // lane half selects K sub-block

    const int nbase = blockIdx.x * BN;
    const int mbase = blockIdx.y * BM;

    // staging: both tiles are 128 rows x 32 cols; 2 threads/row, 16 elems each
    const int st_row = tid >> 1;            // 0..127
    const int st_col = (tid & 1) * 16;      // 0 or 16
    int a_grow = mbase + st_row;
    if (a_grow >= M) a_grow = M - 1;        // clamp tail reads (stores guarded)
    const int b_grow = nbase + st_row;

    const __bf16* gA = A  + (size_t)a_grow * K + st_col;
    const __bf16* gB = Bt + (size_t)b_grow * K + st_col;
    const int lofs = st_row * LDSK + st_col;

    v8f acc[2][4];
    #pragma unroll
    for (int i = 0; i < 2; ++i)
        #pragma unroll
        for (int j = 0; j < 4; ++j) { v8f z = {}; acc[i][j] = z; }

    auto stage = [&](int buf, int kt) {
        cp16_g2l(&sA[buf][lofs],     gA + kt);
        cp16_g2l(&sA[buf][lofs + 8], gA + kt + 8);
        cp16_g2l(&sB[buf][lofs],     gB + kt);
        cp16_g2l(&sB[buf][lofs + 8], gB + kt + 8);
    };

    stage(0, 0);   // prologue

    const int nk = K / BK;
    for (int ik = 0; ik < nk; ++ik) {
        const int  cur      = ik & 1;
        const bool has_next = (ik + 1 < nk);
        if (has_next) stage(cur ^ 1, (ik + 1) * BK);  // overlap DMA with compute
        wait_stage(has_next);                         // current buffer landed
        __syncthreads();

        // fragment loads (ISA 16-bit A layout: lane r/half; K = 16*(v>=4)+8*half+..)
        FragU af[2], bfr[4];
        #pragma unroll
        for (int i = 0; i < 2; ++i) {
            const __bf16* base = &sA[cur][(wm * 32 + i * 16 + r) * LDSK + half * 8];
            af[i].u4[0] = *(const uint4*)(base);
            af[i].u4[1] = *(const uint4*)(base + 16);
        }
        #pragma unroll
        for (int j = 0; j < 4; ++j) {
            const __bf16* base = &sB[cur][(wn * 64 + j * 16 + r) * LDSK + half * 8];
            bfr[j].u4[0] = *(const uint4*)(base);
            bfr[j].u4[1] = *(const uint4*)(base + 16);
        }

        #pragma unroll
        for (int i = 0; i < 2; ++i)
            #pragma unroll
            for (int j = 0; j < 4; ++j)
                acc[i][j] = __builtin_amdgcn_wmma_f32_16x16x32_bf16(
                    /*neg_a=*/false, af[i].v, /*neg_b=*/false, bfr[j].v,
                    /*c_mod=*/(short)0, acc[i][j],
                    /*reuse_a=*/false, /*reuse_b=*/false);

        __syncthreads();   // everyone done reading buf[cur] before it is restaged
    }

    // epilogue: C/D layout -> row = base + half*8 + vgpr, col = base + r
    #pragma unroll
    for (int i = 0; i < 2; ++i) {
        #pragma unroll
        for (int j = 0; j < 4; ++j) {
            const int col = nbase + wn * 64 + j * 16 + r;
            const float bv = bias[col];
            #pragma unroll
            for (int v = 0; v < 8; ++v) {
                const int row = mbase + wm * 32 + i * 16 + half * 8 + v;
                if (row < M) {
                    float val = acc[i][j][v] + bv;
                    if (EPI == 0) {
                        val = 0.5f * val * (1.0f + erff(val * 0.70710678118654752f));
                        ((__bf16*)Cout)[(size_t)row * Nn + col] = (__bf16)val;
                    } else {
                        ((float*)Cout)[(size_t)row * Nn + col] = val;
                    }
                }
            }
        }
    }
}

// ---------------------------------------------------------------------------
// Kernel 4: out[:,0]=y[:,0];  out[:,1+c] = sum_k rot[c,k]*y[:,1+k]
// ---------------------------------------------------------------------------
__global__ __launch_bounds__(128)
void rot_epilogue_kernel(const float* __restrict__ Y, const float* __restrict__ rot,
                         float* __restrict__ Out)
{
    const int n = blockIdx.x;
    const int f = threadIdx.x;
    __shared__ float R[9];
    if (f < 9) R[f] = rot[(size_t)n * 9 + f];
    __syncthreads();

    const float* y = Y + (size_t)n * OUT_DIMS;
    float*       o = Out + (size_t)n * OUT_DIMS;
    o[f] = y[f];
    const float y1 = y[1 * FIBER + f];
    const float y2 = y[2 * FIBER + f];
    const float y3 = y[3 * FIBER + f];
    #pragma unroll
    for (int c = 0; c < 3; ++c)
        o[(1 + c) * FIBER + f] = R[c * 3 + 0] * y1 + R[c * 3 + 1] * y2 + R[c * 3 + 2] * y3;
}

// ---------------------------------------------------------------------------
extern "C" void kernel_launch(void* const* d_in, const int* in_sizes, int n_in,
                              void* d_out, int out_size, void* d_ws, size_t ws_size,
                              hipStream_t stream) {
    const float* x     = (const float*)d_in[0];
    const float* rot   = (const float*)d_in[1];
    const float* extra = (const float*)d_in[2];
    const float* W1    = (const float*)d_in[3];
    const float* b1    = (const float*)d_in[4];
    const float* W2    = (const float*)d_in[5];
    const float* b2    = (const float*)d_in[6];
    float* out = (float*)d_out;

    const int M = in_sizes[0] / (4 * FIBER);   // number of nodes

    // workspace layout (256B aligned). Y (f32, M*512*4) aliases X1 (bf16, M*1152*2).
    char* ws = (char*)d_ws;
    auto align256 = [](size_t v) { return (v + 255) & ~(size_t)255; };
    const size_t offX1  = 0;
    const size_t offH   = align256((size_t)M * IN_DIM * sizeof(__bf16));
    const size_t offW1t = align256(offH + (size_t)M * HIDDEN * sizeof(__bf16));
    const size_t offW2t = align256(offW1t + (size_t)IN_DIM * HIDDEN * sizeof(__bf16));

    __bf16* X1  = (__bf16*)(ws + offX1);
    __bf16* H   = (__bf16*)(ws + offH);
    __bf16* W1t = (__bf16*)(ws + offW1t);
    __bf16* W2t = (__bf16*)(ws + offW2t);
    float*  Y   = (float*)(ws + offX1);        // alias: X1 dead after GEMM1

    // weight transpose+convert (small)
    {
        const int tot1 = IN_DIM * HIDDEN;
        transpose_bf16_kernel<<<(tot1 + 255) / 256, 256, 0, stream>>>(W1, W1t, IN_DIM, HIDDEN);
        const int tot2 = HIDDEN * OUT_DIMS;
        transpose_bf16_kernel<<<(tot2 + 255) / 256, 256, 0, stream>>>(W2, W2t, HIDDEN, OUT_DIMS);
    }

    // input build
    build_x1_kernel<<<M, 128, 0, stream>>>(x, rot, extra, X1);

    // GEMM1: H = gelu(X1 @ W1 + b1)   (M x 1152) x (1152 x 1024)
    {
        dim3 grid(HIDDEN / BN, (M + BM - 1) / BM);
        gemm_wmma_kernel<0><<<grid, 256, 0, stream>>>(X1, W1t, b1, (void*)H, M, HIDDEN, IN_DIM);
    }

    // GEMM2: Y = H @ W2 + b2          (M x 1024) x (1024 x 512)
    {
        dim3 grid(OUT_DIMS / BN, (M + BM - 1) / BM);
        gemm_wmma_kernel<1><<<grid, 256, 0, stream>>>(H, W2t, b2, (void*)Y, M, OUT_DIMS, HIDDEN);
    }

    // rotation epilogue -> out
    rot_epilogue_kernel<<<M, 128, 0, stream>>>(Y, rot, out);
}
